// MultiHeadAttention_29712583753855
// MI455X (gfx1250) — matrix operand order repaired and checked by reading, not compiled, not run on hardware
//
#include <hip/hip_runtime.h>
#include <stdint.h>

// MHA: N=2, T=4096, D=768, H=12, DH=64.  All matmuls on v_wmma_f32_16x16x32_bf16.
// Attention stages K/V tiles into LDS via async global->LDS DMA (ASYNCcnt),
// double-buffered across 64-key tiles.
#define NB 2
#define TS 4096
#define DM 768
#define NH 12
#define DHD 64
#define MR (NB * TS)   // 8192 rows

typedef __bf16 bf16_t;
typedef __bf16 v16bf __attribute__((ext_vector_type(16)));
typedef __bf16 v8bf  __attribute__((ext_vector_type(8)));
typedef float  v8f   __attribute__((ext_vector_type(8)));
typedef int    v4i   __attribute__((vector_size(16)));

#ifndef __has_builtin
#define __has_builtin(x) 0
#endif
#if __has_builtin(__builtin_amdgcn_global_load_async_to_lds_b128)
#define HAVE_ASYNC_LDS 1
#else
#define HAVE_ASYNC_LDS 0
#endif

static __device__ __forceinline__ void wait_async0() {
#if __has_builtin(__builtin_amdgcn_s_wait_asynccnt)
  __builtin_amdgcn_s_wait_asynccnt(0);
#else
  asm volatile("s_wait_asynccnt 0x0" ::: "memory");
#endif
}

static __device__ __forceinline__ v8f wmma_bf16(v16bf a, v16bf b, v8f c) {
  // (neg_a, A, neg_b, B, c_mod, C, reuse_a, reuse_b)
  return __builtin_amdgcn_wmma_f32_16x16x32_bf16(false, a, false, b, (short)0, c,
                                                 false, false);
}

// A-matrix 16x32 bf16 fragment (row-major source). k order chosen here (not the ISA table order):
// lanes 0-15 hold K 0..15, lanes 16-31 hold K 16..31 (contiguous, matches B).
static __device__ __forceinline__ v16bf frag_a(const bf16_t* row, int k0, int laneHi) {
  const bf16_t* p = row + k0 + (laneHi ? 16 : 0);
  v8bf lo = *(const v8bf*)(p);
  v8bf hi = *(const v8bf*)(p + 8);
  return __builtin_shufflevector(lo, hi, 0, 1, 2, 3, 4, 5, 6, 7,
                                 8, 9, 10, 11, 12, 13, 14, 15);
}
// B-matrix 32x16 bf16 fragment: lane L<16 holds column N=L, K 0..15 contiguous;
// lane L>=16 holds column N=L-16, K 16..31.  B^T row == row of (W / K / V^T),
// so a B fragment is one 32-byte contiguous load from a row-major B^T source.

// Stage one 64x64 bf16 tile (8KB) from global (row stride gStride elems) into
// LDS, cooperatively by 256 threads: 512 x 16B segments, 2 per thread.
static __device__ __forceinline__ void stage_tile64(const bf16_t* gsrc,
                                                    size_t gStride,
                                                    bf16_t* lds, int tid) {
#pragma unroll
  for (int s = 0; s < 2; ++s) {
    const int seg = tid + s * 256;      // 0..511
    const int r = seg >> 3;             // row 0..63
    const int c = (seg & 7) * 8;        // bf16 column, 16B segment
    const bf16_t* gp = gsrc + (size_t)r * gStride + c;
    bf16_t* lp = lds + r * 64 + c;
#if HAVE_ASYNC_LDS
    typedef __attribute__((address_space(1))) v4i* gv4p;
    typedef __attribute__((address_space(3))) v4i* lv4p;
    __builtin_amdgcn_global_load_async_to_lds_b128((gv4p)(void*)gp, (lv4p)(void*)lp,
                                                   0, 0);
#else
    *(v8bf*)lp = *(const v8bf*)gp;
#endif
  }
}

// ---------------------------------------------------------------- convert ---
__global__ void mha_cvt_bf16(const float* __restrict__ x,
                             const float* __restrict__ wq, const float* __restrict__ wk,
                             const float* __restrict__ wv, const float* __restrict__ wo,
                             bf16_t* __restrict__ xb,
                             bf16_t* __restrict__ wqb, bf16_t* __restrict__ wkb,
                             bf16_t* __restrict__ wvb, bf16_t* __restrict__ wob) {
  const int stride = gridDim.x * blockDim.x;
  const int tid = blockIdx.x * blockDim.x + threadIdx.x;
  for (int i = tid; i < MR * DM; i += stride) xb[i] = (bf16_t)x[i];
  for (int i = tid; i < DM * DM; i += stride) {
    wqb[i] = (bf16_t)wq[i];
    wkb[i] = (bf16_t)wk[i];
    wvb[i] = (bf16_t)wv[i];
    wob[i] = (bf16_t)wo[i];
  }
}

// ------------------------------------------------------------------- GEMM ---
// C[M=8192, N=768] = A(bf16) @ W^T + bias.  Block tile 128x128, 8 waves,
// each wave 32x64 (2x4 of 16x16 WMMA tiles), K-loop step 32.
// MODE 0: write bf16 [n,h,t,dh] (Q/K; Q pre-scaled by 1/sqrt(dh))
// MODE 1: write bf16 [n,h,dh,t] (V transposed for the P@V B-operand)
// MODE 2: write f32 row-major (final output)
template <int MODE>
__global__ __launch_bounds__(256) void mha_gemm_xwt(const bf16_t* __restrict__ A,
                                                    const bf16_t* __restrict__ W,
                                                    const float* __restrict__ bias,
                                                    float scale,
                                                    void* __restrict__ dst) {
  const int lane   = threadIdx.x & 31;
  const int wave   = threadIdx.x >> 5;
  const int laneHi = lane >> 4;
  const int lrow   = lane & 15;
  const int m0 = blockIdx.x * 128 + (wave & 3) * 32;
  const int n0 = blockIdx.y * 128 + (wave >> 2) * 64;

  v8f acc[2][4] = {};

  const bf16_t* aRow0 = A + (size_t)(m0 + lrow) * DM;
  const bf16_t* aRow1 = aRow0 + (size_t)16 * DM;
  const bf16_t* bRow[4];
#pragma unroll
  for (int nt = 0; nt < 4; ++nt)
    bRow[nt] = W + (size_t)(n0 + nt * 16 + lrow) * DM + (laneHi ? 16 : 0);

  for (int k0 = 0; k0 < DM; k0 += 32) {
    v16bf a0 = frag_a(aRow0, k0, laneHi);
    v16bf a1 = frag_a(aRow1, k0, laneHi);
#pragma unroll
    for (int nt = 0; nt < 4; ++nt) {
      v16bf b = *(const v16bf*)(bRow[nt] + k0);
      acc[0][nt] = wmma_bf16(a0, b, acc[0][nt]);
      acc[1][nt] = wmma_bf16(a1, b, acc[1][nt]);
    }
  }

  // C layout: lane holds column j = n0+nt*16+(lane&15); element e is row
  // m0+mt*16+e+8*(lane>=16).
#pragma unroll
  for (int mt = 0; mt < 2; ++mt) {
#pragma unroll
    for (int nt = 0; nt < 4; ++nt) {
      const int j = n0 + nt * 16 + lrow;
      const float bj = bias[j];
#pragma unroll
      for (int e = 0; e < 8; ++e) {
        const int m = m0 + mt * 16 + e + laneHi * 8;
        float v = acc[mt][nt][e] + bj;
        if (MODE == 2) {
          ((float*)dst)[(size_t)m * DM + j] = v;
        } else {
          const int n = m >> 12, t = m & (TS - 1);
          const int h = j >> 6, dh = j & 63;
          v *= scale;
          bf16_t* o = (bf16_t*)dst;
          if (MODE == 0)
            o[(((size_t)n * NH + h) * TS + t) * DHD + dh] = (bf16_t)v;
          else
            o[(((size_t)n * NH + h) * DHD + dh) * TS + t] = (bf16_t)v;
        }
      }
    }
  }
}

// -------------------------------------------------------- flash attention ---
// Grid: (T/128, N*H).  8 waves/block; each wave: 16 q-rows x 64 dh output.
// K/V tiles of 64 keys are staged to LDS once per block via async global->LDS
// DMA, double-buffered; waves run S = Q K^T, online softmax, O += P V on WMMA.
__global__ __launch_bounds__(256) void mha_flash_attn(const bf16_t* __restrict__ qb,
                                                      const bf16_t* __restrict__ kbuf,
                                                      const bf16_t* __restrict__ vt,
                                                      const uint8_t* __restrict__ mask,
                                                      bf16_t* __restrict__ ab) {
  __shared__ bf16_t kT[2][64 * 64];    // [key][dh]   8KB per buffer
  __shared__ bf16_t vT[2][64 * 64];    // [dh][key]   8KB per buffer
  __shared__ bf16_t pshm[8][16 * 64];  // per-wave P tile (C->A relayout), 16KB

  const int tid    = threadIdx.x;
  const int lane   = tid & 31;
  const int wave   = tid >> 5;
  const int laneHi = lane >> 4;
  const int lrow   = lane & 15;
  const int bh = blockIdx.y;           // n*NH + h
  const int n  = bh / NH;
  const int h  = bh - n * NH;
  const int q0 = blockIdx.x * 128 + wave * 16;

  const bf16_t* kBase = kbuf + (size_t)bh * TS * DHD;
  const bf16_t* vBase = vt + (size_t)bh * DHD * TS;
  const uint8_t* mBase = mask + (size_t)n * TS;

  // Q fragments (dh split into two K=32 chunks); Q already holds 1/sqrt(dh).
  const bf16_t* qRow = qb + ((size_t)bh * TS + q0 + lrow) * DHD;
  v16bf aQ0 = frag_a(qRow, 0, laneHi);
  v16bf aQ1 = frag_a(qRow, 32, laneHi);

  float rmax[8], rsum[8];
#pragma unroll
  for (int e = 0; e < 8; ++e) { rmax[e] = -1e30f; rsum[e] = 0.f; }
  v8f accO[4] = {};
  bf16_t* pW = &pshm[wave][0];

  // Block-uniform "entire 64-key tile is masked" test (each lane checks 2 keys).
  auto tileMasked = [&](int kb) -> bool {
    const bool m = (mBase[kb + lane] != 0) && (mBase[kb + 32 + lane] != 0);
    return __all(m) != 0;
  };

  bool curMasked = tileMasked(0);
  if (!curMasked) {
    stage_tile64(kBase, DHD, &kT[0][0], tid);        // K rows: [key][dh]
    stage_tile64(vBase, TS, &vT[0][0], tid);         // V^T rows: [dh][key]
  }

  for (int kb = 0, it = 0; kb < TS; kb += 64, ++it) {
    const int buf = it & 1;
    const bool nextMasked = (kb + 64 < TS) ? tileMasked(kb + 64) : true;

    wait_async0();        // this wave's stage requests for `buf` are complete
    __syncthreads();      // all waves' requests complete -> tile readable

    if (!nextMasked) {    // overlap next tile DMA with compute on current
      stage_tile64(kBase + (size_t)(kb + 64) * DHD, DHD, &kT[buf ^ 1][0], tid);
      stage_tile64(vBase + (kb + 64), TS, &vT[buf ^ 1][0], tid);
    }

    if (!curMasked) {
      // S(16x64) = Q(16x64) @ K_tile^T : four 16x16 column tiles, 2 K-steps.
      v8f s[4] = {};
#pragma unroll
      for (int j = 0; j < 4; ++j) {
        const bf16_t* kRow = &kT[buf][(j * 16 + lrow) * 64 + (laneHi ? 16 : 0)];
        s[j] = wmma_bf16(aQ0, *(const v16bf*)(kRow), s[j]);
        s[j] = wmma_bf16(aQ1, *(const v16bf*)(kRow + 32), s[j]);
      }
      // Key-padding mask: lane owns column key = kb + j*16 + lrow.
#pragma unroll
      for (int j = 0; j < 4; ++j) {
        if (mBase[kb + j * 16 + lrow]) {
#pragma unroll
          for (int e = 0; e < 8; ++e) s[j][e] = -1e30f;
        }
      }

      // Online softmax; row reductions across the 16-lane halves (shfl_xor).
#pragma unroll
      for (int e = 0; e < 8; ++e) {
        float mx = fmaxf(fmaxf(s[0][e], s[1][e]), fmaxf(s[2][e], s[3][e]));
        mx = fmaxf(mx, __shfl_xor(mx, 1));
        mx = fmaxf(mx, __shfl_xor(mx, 2));
        mx = fmaxf(mx, __shfl_xor(mx, 4));
        mx = fmaxf(mx, __shfl_xor(mx, 8));
        const float nm = fmaxf(rmax[e], mx);
        const float f = __expf(rmax[e] - nm);
        rmax[e] = nm;
        float sum = 0.f;
#pragma unroll
        for (int j = 0; j < 4; ++j) {
          const float p = __expf(s[j][e] - nm);
          s[j][e] = p;
          sum += p;
        }
        sum += __shfl_xor(sum, 1);
        sum += __shfl_xor(sum, 2);
        sum += __shfl_xor(sum, 4);
        sum += __shfl_xor(sum, 8);
        rsum[e] = rsum[e] * f + sum;
#pragma unroll
        for (int nt = 0; nt < 4; ++nt) accO[nt][e] *= f;
      }

      // Re-layout P (16x64) from C-layout to A-layout via per-wave LDS.
#pragma unroll
      for (int e = 0; e < 8; ++e) {
        const int r = e + laneHi * 8;
#pragma unroll
        for (int j = 0; j < 4; ++j)
          pW[r * 64 + j * 16 + lrow] = (bf16_t)s[j][e];
      }
      asm volatile("s_wait_dscnt 0x0" ::: "memory");  // per-wave LDS RAW fence

      const bf16_t* pRow = pW + lrow * 64;
      v16bf pa0 = frag_a(pRow, 0, laneHi);   // keys 0..31 of tile
      v16bf pa1 = frag_a(pRow, 32, laneHi);  // keys 32..63

      // O(16x64) += P(16x64) @ V_tile(64x64); B^T rows are V^T rows in LDS.
#pragma unroll
      for (int nt = 0; nt < 4; ++nt) {
        const bf16_t* vRow = &vT[buf][(nt * 16 + lrow) * 64 + (laneHi ? 16 : 0)];
        accO[nt] = wmma_bf16(pa0, *(const v16bf*)(vRow), accO[nt]);
        accO[nt] = wmma_bf16(pa1, *(const v16bf*)(vRow + 32), accO[nt]);
      }
    }

    curMasked = nextMasked;
    __syncthreads();  // all waves done reading `buf` before it is re-staged
  }

  // Normalize and store bf16 attention output at [n*TS+t][h*64+dh].
#pragma unroll
  for (int e = 0; e < 8; ++e) {
    const float inv = 1.f / rsum[e];
    const int t = q0 + e + laneHi * 8;
    const size_t rowOff = ((size_t)n * TS + t) * DM + h * DHD;
#pragma unroll
    for (int nt = 0; nt < 4; ++nt)
      ab[rowOff + nt * 16 + lrow] = (bf16_t)(accO[nt][e] * inv);
  }
}

// ------------------------------------------------------------------ launch ---
extern "C" void kernel_launch(void* const* d_in, const int* in_sizes, int n_in,
                              void* d_out, int out_size, void* d_ws, size_t ws_size,
                              hipStream_t stream) {
  const float*   x    = (const float*)d_in[0];
  const uint8_t* mask = (const uint8_t*)d_in[1];  // jnp bool -> 1 byte/elem
  const float*   Wq   = (const float*)d_in[2];
  const float*   bq   = (const float*)d_in[3];
  const float*   Wk   = (const float*)d_in[4];
  const float*   bk   = (const float*)d_in[5];
  const float*   Wv   = (const float*)d_in[6];
  const float*   bv   = (const float*)d_in[7];
  const float*   Wo   = (const float*)d_in[8];
  const float*   bo   = (const float*)d_in[9];
  float* out = (float*)d_out;

  char* ws = (char*)d_ws;
  size_t off = 0;
  auto alloc = [&](size_t bytes) -> void* {
    void* p = ws + off;
    off += (bytes + 255) & ~(size_t)255;
    return p;
  };
  bf16_t* xb  = (bf16_t*)alloc((size_t)MR * DM * sizeof(bf16_t));  // also reused as attn out
  bf16_t* wqb = (bf16_t*)alloc((size_t)DM * DM * sizeof(bf16_t));
  bf16_t* wkb = (bf16_t*)alloc((size_t)DM * DM * sizeof(bf16_t));
  bf16_t* wvb = (bf16_t*)alloc((size_t)DM * DM * sizeof(bf16_t));
  bf16_t* wob = (bf16_t*)alloc((size_t)DM * DM * sizeof(bf16_t));
  bf16_t* qbf = (bf16_t*)alloc((size_t)NB * NH * TS * DHD * sizeof(bf16_t));
  bf16_t* kbf = (bf16_t*)alloc((size_t)NB * NH * TS * DHD * sizeof(bf16_t));
  bf16_t* vtb = (bf16_t*)alloc((size_t)NB * NH * DHD * TS * sizeof(bf16_t));
  // total workspace ~= 55 MB

  mha_cvt_bf16<<<2048, 256, 0, stream>>>(x, Wq, Wk, Wv, Wo, xb, wqb, wkb, wvb, wob);

  dim3 gG(MR / 128, DM / 128);  // 64 x 6
  mha_gemm_xwt<0><<<gG, 256, 0, stream>>>(xb, wqb, bq, 0.125f, qbf);  // Q * 1/sqrt(64)
  mha_gemm_xwt<0><<<gG, 256, 0, stream>>>(xb, wkb, bk, 1.0f, kbf);
  mha_gemm_xwt<1><<<gG, 256, 0, stream>>>(xb, wvb, bv, 1.0f, vtb);    // V transposed

  dim3 gA(TS / 128, NB * NH);   // 32 x 24
  mha_flash_attn<<<gA, 256, 0, stream>>>(qbf, kbf, vtb, mask, xb /* attn out, aliases xb */);

  mha_gemm_xwt<2><<<gG, 256, 0, stream>>>(xb, wob, bo, 1.0f, out);
}
